// AdaptConv2d_28269474742495
// MI455X (gfx1250) — matrix operand order repaired and checked by reading, not compiled, not run on hardware
//
#include <hip/hip_runtime.h>
#include <math.h>

// CDNA5 / gfx1250: wave32, WMMA bf16 16x16x32 with f32 accumulate.
typedef __attribute__((ext_vector_type(16))) __bf16        v16bf;
typedef __attribute__((ext_vector_type(8)))  float         v8f;

struct Q2 { uint4 a, b; };   // 32 bytes == one v16bf fragment

#define B_    32
#define C_    256
#define O_    256
#define HW_   3136
#define HID_  16

// ---- fp32 <-> bf16 helpers (round-to-nearest-even) -------------------------
__device__ __forceinline__ unsigned short f2bf(float f) {
    unsigned int u = __builtin_bit_cast(unsigned int, f);
    u += 0x7fffu + ((u >> 16) & 1u);
    return (unsigned short)(u >> 16);
}
__device__ __forceinline__ float bf2f(unsigned short h) {
    unsigned int u = ((unsigned int)h) << 16;
    return __builtin_bit_cast(float, u);
}
__device__ __forceinline__ unsigned int pack_split(float a0, float a1,
                                                   unsigned int& lo_out) {
    unsigned short h0 = f2bf(a0), h1 = f2bf(a1);
    unsigned short g0 = f2bf(a0 - bf2f(h0));
    unsigned short g1 = f2bf(a1 - bf2f(h1));
    lo_out = (unsigned int)g0 | ((unsigned int)g1 << 16);
    return (unsigned int)h0 | ((unsigned int)h1 << 16);
}

// ---- K1: global average pool over HW ---------------------------------------
__global__ __launch_bounds__(256) void pool_kernel(const float* __restrict__ x,
                                                   float* __restrict__ pooled) {
    __shared__ float red[256];
    const int bc = blockIdx.x;                 // b*C_ + c
    const float* p = x + (size_t)bc * HW_;
    float s = 0.f;
    for (int i = threadIdx.x; i < HW_; i += 256) s += p[i];
    red[threadIdx.x] = s;
    __syncthreads();
    for (int off = 128; off > 0; off >>= 1) {
        if (threadIdx.x < off) red[threadIdx.x] += red[threadIdx.x + off];
        __syncthreads();
    }
    if (threadIdx.x == 0) pooled[bc] = red[0] * (1.0f / (float)HW_);
}

// ---- K2: hid = silu(pooled @ w_se1^T) --------------------------------------
__global__ __launch_bounds__(256) void se_kernel(const float* __restrict__ pooled,
                                                 const float* __restrict__ w_se1,
                                                 float* __restrict__ hid) {
    int i = blockIdx.x * 256 + threadIdx.x;    // < B_*HID_
    if (i >= B_ * HID_) return;
    const int b = i / HID_, h = i % HID_;
    const float* pr = pooled + b * C_;
    const float* wr = w_se1 + h * C_;
    float s = 0.f;
    for (int c = 0; c < C_; ++c) s += pr[c] * wr[c];
    hid[i] = s / (1.0f + __expf(-s));          // SiLU
}

// ---- K3: w_b = w_conv + hid @ w_se_out^T, split to bf16 hi/lo --------------
__global__ __launch_bounds__(256) void wb_kernel(const float* __restrict__ w_conv,
                                                 const float* __restrict__ w_se_out,
                                                 const float* __restrict__ hid,
                                                 unsigned short* __restrict__ whi,
                                                 unsigned short* __restrict__ wlo) {
    size_t i = (size_t)blockIdx.x * 256 + threadIdx.x;   // < B_*O_*C_ (2M)
    const int b    = (int)(i >> 16);                     // O_*C_ == 65536
    const int rest = (int)(i & 65535);                   // o*C_ + c
    const float* hr = hid + b * HID_;
    const float* wr = w_se_out + (size_t)rest * HID_;
    float s = w_conv[rest];
#pragma unroll
    for (int j = 0; j < HID_; ++j) s += hr[j] * wr[j];
    unsigned short h = f2bf(s);
    whi[i] = h;
    wlo[i] = f2bf(s - bf2f(h));
}

// ---- K4: out[b] = w_b[b] (O x C) @ x[b] (C x HW) via bf16 WMMA -------------
// grid = (HW_/64, O_/128, B_), block = 256 (8 waves).
// Block tile: 128(M) x 64(N). Wave w owns M-tile w and all 4 N-tiles
// (register tiling: 4 accumulators -> each A fragment feeds 4 WMMA triples,
// cutting A-side L2 traffic 4x). The x tile is staged in LDS PRE-SWIZZLED
// into WMMA B-fragment order (frag[ntile][kblock][lane][dword], packed bf16
// pairs, hi & lo), so the steady-state loop is pure b128 loads + v_wmma.
__global__ __launch_bounds__(256, 2) void adapt_gemm(const float* __restrict__ x,
                                                     const unsigned short* __restrict__ whi,
                                                     const unsigned short* __restrict__ wlo,
                                                     float* __restrict__ out) {
    // 4 ntiles x 8 kblocks x 32 lanes x 8 dwords = 8192 u32 per array (32 KB)
    __shared__ unsigned int fragBhi[8192];
    __shared__ unsigned int fragBlo[8192];

    const int n0     = blockIdx.x * 64;
    const int m_base = blockIdx.y * 128;
    const int b      = blockIdx.z;
    const int t      = threadIdx.x;
    const int lane   = t & 31;
    const int wave   = t >> 5;

    // ---- stage + convert + swizzle the x tile [C_=256 K x 64 N] ------------
    // thread t handles K-pair row p = t>>1 (K = 2p, 2p+1), a 32-column half
    // (nh = 0 or 32), processed as two 16-column chunks (one ntile each).
    {
        const int p  = t >> 1;                 // 0..127
        const int nh = (t & 1) * 32;           // 0 or 32
        const int kb = p >> 4;
        const int hi = (p >> 2) & 1;
        const int d  = (p & 3) | (((p >> 3) & 1) << 2);
        const float* r0 = x + ((size_t)b * C_ + 2 * p) * HW_ + n0 + nh;
        const float* r1 = r0 + HW_;
#pragma unroll
        for (int ch = 0; ch < 2; ++ch) {
            const int ntile = (nh >> 4) + ch;  // constant per chunk
            const float4* q0 = (const float4*)(r0 + ch * 16);
            const float4* q1 = (const float4*)(r1 + ch * 16);
            float4 e0 = q0[0], e1 = q0[1], e2 = q0[2], e3 = q0[3];
            float4 g0 = q1[0], g1 = q1[1], g2 = q1[2], g3 = q1[3];
            float ra[16] = {e0.x, e0.y, e0.z, e0.w, e1.x, e1.y, e1.z, e1.w,
                            e2.x, e2.y, e2.z, e2.w, e3.x, e3.y, e3.z, e3.w};
            float rb[16] = {g0.x, g0.y, g0.z, g0.w, g1.x, g1.y, g1.z, g1.w,
                            g2.x, g2.y, g2.z, g2.w, g3.x, g3.y, g3.z, g3.w};
#pragma unroll
            for (int j = 0; j < 16; ++j) {
                const int l   = hi * 16 + j;   // fragment lane
                const int idx = ntile * 2048 + kb * 256 + l * 8 + d;
                unsigned int lo;
                unsigned int hi32 = pack_split(ra[j], rb[j], lo);
                fragBhi[idx] = hi32;
                fragBlo[idx] = lo;
            }
        }
    }
    __syncthreads();

    const int l15  = lane & 15;
    const int hi16 = lane >> 4;                // 0 for lanes 0-15, 1 for 16-31
    const int m    = m_base + wave * 16 + l15; // A-matrix row for this lane

    const unsigned int* arow_hi =
        (const unsigned int*)(whi + ((size_t)b * O_ + m) * C_);
    const unsigned int* arow_lo =
        (const unsigned int*)(wlo + ((size_t)b * O_ + m) * C_);

    v8f acc[4] = {{}, {}, {}, {}};

#pragma unroll
    for (int kb = 0; kb < 8; ++kb) {
        const int k0h = kb * 16;               // K-pair base = k0/2
        // A fragment: dwords 0-3 at pair idx k0h + hi16*4, dwords 4-7 at +8.
        const uint4* Ah = (const uint4*)(arow_hi + k0h + hi16 * 4);
        const uint4* Al = (const uint4*)(arow_lo + k0h + hi16 * 4);
        Q2 qa_hi; qa_hi.a = Ah[0]; qa_hi.b = Ah[2];
        Q2 qa_lo; qa_lo.a = Al[0]; qa_lo.b = Al[2];
        const v16bf A_hi = __builtin_bit_cast(v16bf, qa_hi);
        const v16bf A_lo = __builtin_bit_cast(v16bf, qa_lo);

#pragma unroll
        for (int nt = 0; nt < 4; ++nt) {
            // B fragment: 8 contiguous dwords per lane (two ds_load_b128).
            const uint4* Bh = (const uint4*)&fragBhi[nt * 2048 + kb * 256 + lane * 8];
            const uint4* Bl = (const uint4*)&fragBlo[nt * 2048 + kb * 256 + lane * 8];
            Q2 qb_hi; qb_hi.a = Bh[0]; qb_hi.b = Bh[1];
            Q2 qb_lo; qb_lo.a = Bl[0]; qb_lo.b = Bl[1];
            const v16bf B_hi = __builtin_bit_cast(v16bf, qb_hi);
            const v16bf B_lo = __builtin_bit_cast(v16bf, qb_lo);

            // 2-term bf16 decomposition: hi*hi + lo*hi + hi*lo
            acc[nt] = __builtin_amdgcn_wmma_f32_16x16x32_bf16(
                false, A_hi, false, B_hi, (short)0, acc[nt], false, false);
            acc[nt] = __builtin_amdgcn_wmma_f32_16x16x32_bf16(
                false, A_lo, false, B_hi, (short)0, acc[nt], false, false);
            acc[nt] = __builtin_amdgcn_wmma_f32_16x16x32_bf16(
                false, A_hi, false, B_lo, (short)0, acc[nt], false, false);
        }
    }

    // C/D layout: VGPR r -> M = r (lanes 0-15) / r+8 (lanes 16-31), N = lane&15.
    float* obase = out + (size_t)b * O_ * HW_ + n0 + l15;
#pragma unroll
    for (int nt = 0; nt < 4; ++nt) {
#pragma unroll
        for (int r = 0; r < 8; ++r) {
            const int M = m_base + wave * 16 + r + hi16 * 8;
            obase[(size_t)M * HW_ + nt * 16] = acc[nt][r];
        }
    }
}

// ---- launcher ---------------------------------------------------------------
extern "C" void kernel_launch(void* const* d_in, const int* in_sizes, int n_in,
                              void* d_out, int out_size, void* d_ws, size_t ws_size,
                              hipStream_t stream) {
    const float* x        = (const float*)d_in[0];  // [32,256,56,56]
    const float* w_conv   = (const float*)d_in[1];  // [256,256,1,1]
    const float* w_se1    = (const float*)d_in[2];  // [16,256]
    const float* w_se_out = (const float*)d_in[3];  // [65536,16]
    float* out = (float*)d_out;                     // [32,256,56,56]

    // workspace layout: pooled(32K B) | hid(2K B) | whi(4 MB) | wlo(4 MB)
    float* pooled = (float*)d_ws;
    float* hid    = pooled + B_ * C_;
    unsigned short* whi = (unsigned short*)(hid + B_ * HID_);
    unsigned short* wlo = whi + (size_t)B_ * O_ * C_;

    pool_kernel<<<B_ * C_, 256, 0, stream>>>(x, pooled);
    se_kernel<<<(B_ * HID_ + 255) / 256, 256, 0, stream>>>(pooled, w_se1, hid);
    wb_kernel<<<(B_ * O_ * C_) / 256, 256, 0, stream>>>(w_conv, w_se_out, hid, whi, wlo);

    dim3 grid(HW_ / 64, O_ / 128, B_);              // (49, 2, 32)
    adapt_gemm<<<grid, 256, 0, stream>>>(x, whi, wlo, out);
}
